// MultiHeadAttention_10909216931929
// MI455X (gfx1250) — matrix operand order repaired and checked by reading, not compiled, not run on hardware
//
#include <hip/hip_runtime.h>
#include <cstdint>
#include <cstddef>

// ---------------------------------------------------------------------------
// MHA for MI455X (gfx1250): all 5 matmul stages via v_wmma_f32_16x16x32_bf16.
// GEMM B-tiles are DMA'd into LDS by the Tensor Data Mover (tensor_load_to_lds
// + s_wait_tensorcnt), with TDM padding generating the conflict-free stride-40
// LDS layout. Softmax is fp32 VALU with b128 LDS accesses.
// ---------------------------------------------------------------------------

typedef __bf16 bf16;
typedef __attribute__((ext_vector_type(16))) __bf16 v16bf;
typedef __attribute__((ext_vector_type(8)))  float  v8f;
typedef __attribute__((ext_vector_type(4)))  unsigned int u32x4;
typedef __attribute__((ext_vector_type(8)))  unsigned int u32x8;
typedef __attribute__((ext_vector_type(4)))  float  f32x4;

constexpr int Bb = 8, Ss = 1024, Dd = 1024, Hh = 16, DHh = 64;

union Frag16 { v16bf v; u32x4 u[2]; };
union Pack8  { bf16  h[8]; u32x4 u; };

__device__ inline v8f wmma_bf16(v16bf a, v16bf b, v8f c) {
  // 8 args: (neg_a, A, neg_b, B, c_mod, C, reuse_a, reuse_b)
  return __builtin_amdgcn_wmma_f32_16x16x32_bf16(false, a, false, b,
                                                 (short)0, c, false, false);
}

__device__ inline float wave_max(float v) {
  #pragma unroll
  for (int m = 16; m >= 1; m >>= 1) v = fmaxf(v, __shfl_xor(v, m, 32));
  return v;
}
__device__ inline float wave_sum(float v) {
  #pragma unroll
  for (int m = 16; m >= 1; m >>= 1) v += __shfl_xor(v, m, 32);
  return v;
}

// ---------------------------------------------------------------------------
// TDM: DMA a 64-row x 32-col bf16 tile (row stride K elements) from global to
// LDS. pad_enable with interval=64B, amount=16B writes rows at stride 40 bf16
// (80B) -- identical to the hand-padded conflict-free layout.
// Descriptor packing per CDNA5 ISA.
// ---------------------------------------------------------------------------
__device__ inline void tdm_load_btile64x32(const bf16* gsrc, unsigned lds_off,
                                           unsigned row_stride_elems) {
  const unsigned long long ga = (unsigned long long)(uintptr_t)gsrc;
  u32x4 g0;
  g0[0] = 1u;                                   // count=1, user mode
  g0[1] = lds_off;                              // lds_addr (bytes)
  g0[2] = (unsigned)ga;                         // global_addr[31:0]
  g0[3] = (unsigned)((ga >> 32) & 0x1FFFFFFu)   // global_addr[56:32]
          | 0x80000000u;                        // type=2 ("image")
  u32x8 g1;
  g1[0] = (1u << 16)                            // data_size = 1 -> 2 bytes
        | (1u << 20)                            // pad_enable
        | (3u << 22)                            // pad_interval: 16 DW = 64B
        | (3u << 25);                           // pad_amount:   4 DW = 16B
  g1[1] = 0u;                                   // abar addr=0, td0[15:0]=0
  g1[2] = 1u;                                   // td0=65536, td1[15:0]=0
  g1[3] = 1u | (32u << 16);                     // td1=65536, tile_dim0=32
  g1[4] = 64u;                                  // tile_dim1=64, tile_dim2=0
  g1[5] = row_stride_elems;                     // tensor_dim0_stride lo
  g1[6] = 0u;                                   // stride hi / dim1_stride lo
  g1[7] = 0u;                                   // dim1_stride hi
  asm volatile("tensor_load_to_lds %0, %1" :: "s"(g0), "s"(g1) : "memory");
}

// ---------------------------------------------------------------------------
// Weight transpose + fp32->bf16: Wt[n][k] = (bf16)W[k][n]
// ---------------------------------------------------------------------------
__global__ __launch_bounds__(256)
void transpose_to_bf16(const float* __restrict__ W, bf16* __restrict__ Wt,
                       int K, int N) {
  __shared__ float tile[32][33];
  const int tx = threadIdx.x & 31;
  const int ty = threadIdx.x >> 5;        // 0..7
  const int n0 = blockIdx.x * 32;
  const int k0 = blockIdx.y * 32;
  #pragma unroll
  for (int j = ty; j < 32; j += 8)
    tile[j][tx] = W[(size_t)(k0 + j) * N + n0 + tx];
  __syncthreads();
  #pragma unroll
  for (int j = ty; j < 32; j += 8)
    Wt[(size_t)(n0 + j) * K + k0 + tx] = (bf16)tile[tx][j];
}

// ---------------------------------------------------------------------------
// C[M,N] = A[M,K] x Bt[N,K]^T.  A = fp32 (converted inline) or bf16.
// Block: 256 threads = 8 waves; tile 128(M) x 64(N); B tile via TDM into LDS.
// OUT_MODE 0: bf16 row-major  1: bf16 transposed per batch of Ss rows
//          2: fp32 row-major
// ---------------------------------------------------------------------------
template<typename AT, int OUT_MODE>
__global__ __launch_bounds__(256)
void gemm_bf16_wmma(const AT* __restrict__ A, const bf16* __restrict__ Bt,
                    void* __restrict__ Cout, int M, int N, int K) {
  __shared__ __align__(16) bf16 Bs[64 * 40];  // TDM-padded: stride 40 bf16
  const int tid  = threadIdx.x;
  const int lane = tid & 31, wave = tid >> 5;
  const int ln = lane & 15, hi = lane >> 4;
  const int aoff = hi * 8, boff = hi * 8, coff = hi * 8;
  const int m0 = blockIdx.x * 128 + wave * 16;
  const int n0 = blockIdx.y * 64;
  const unsigned ldsBs = (unsigned)(uintptr_t)(&Bs[0]);  // LDS byte offset

  v8f acc[4];
  #pragma unroll
  for (int t = 0; t < 4; ++t)
    #pragma unroll
    for (int r = 0; r < 8; ++r) acc[t][r] = 0.0f;

  for (int k0 = 0; k0 < K; k0 += 32) {
    __syncthreads();                       // all reads of Bs done
    if (wave == 0) {
      tdm_load_btile64x32(Bt + (size_t)n0 * K + k0, ldsBs, (unsigned)K);
      __builtin_amdgcn_s_wait_tensorcnt(0);
    }
    __syncthreads();                       // tile visible to all waves

    Frag16 a;
    const AT* arow = A + (size_t)(m0 + ln) * K + k0;
    if constexpr (sizeof(AT) == 4) {      // fp32 A: load + convert to bf16
      f32x4 x0 = *(const f32x4*)(arow + aoff);
      f32x4 x1 = *(const f32x4*)(arow + aoff + 4);
      f32x4 x2 = *(const f32x4*)(arow + aoff + 16);
      f32x4 x3 = *(const f32x4*)(arow + aoff + 20);
      #pragma unroll
      for (int i = 0; i < 4; ++i) {
        a.v[i]      = (bf16)x0[i];
        a.v[4 + i]  = (bf16)x1[i];
        a.v[8 + i]  = (bf16)x2[i];
        a.v[12 + i] = (bf16)x3[i];
      }
    } else {                              // bf16 A: two b128 loads
      a.u[0] = *(const u32x4*)(arow + aoff);
      a.u[1] = *(const u32x4*)(arow + 16 + aoff);
    }

    #pragma unroll
    for (int t = 0; t < 4; ++t) {
      Frag16 bfr;
      const bf16* bsrow = &Bs[(t * 16 + ln) * 40 + boff];
      bfr.u[0] = *(const u32x4*)(bsrow);
      bfr.u[1] = *(const u32x4*)(bsrow + 16);
      acc[t] = wmma_bf16(a.v, bfr.v, acc[t]);
    }
  }

  if constexpr (OUT_MODE == 0) {
    bf16* C = (bf16*)Cout;
    #pragma unroll
    for (int t = 0; t < 4; ++t)
      #pragma unroll
      for (int r = 0; r < 8; ++r)
        C[(size_t)(m0 + coff + r) * N + n0 + t * 16 + ln] = (bf16)acc[t][r];
  } else if constexpr (OUT_MODE == 2) {
    float* C = (float*)Cout;
    #pragma unroll
    for (int t = 0; t < 4; ++t)
      #pragma unroll
      for (int r = 0; r < 8; ++r)
        C[(size_t)(m0 + coff + r) * N + n0 + t * 16 + ln] = acc[t][r];
  } else {                                // transposed-per-batch bf16 (for V)
    bf16* C = (bf16*)Cout;
    const int bb = m0 / Ss;
    const int s0 = (m0 % Ss) + coff;      // 8-aligned -> 16B-aligned store
    #pragma unroll
    for (int t = 0; t < 4; ++t) {
      Pack8 p;
      #pragma unroll
      for (int r = 0; r < 8; ++r) p.h[r] = (bf16)acc[t][r];
      *(u32x4*)(C + ((size_t)bb * Dd + n0 + t * 16 + ln) * (size_t)Ss + s0) = p.u;
    }
  }
}

// ---------------------------------------------------------------------------
// Attention: one block per (b, h, 32 q-rows). Scores via WMMA into LDS (bf16,
// scaled + valid_len masked), fp32 softmax (vectorized b128 LDS access), then
// P@V via WMMA with A-frags from LDS and B-frags from transposed V.
// ---------------------------------------------------------------------------
__global__ __launch_bounds__(256)
void attention_kernel(const bf16* __restrict__ Qp, const bf16* __restrict__ Kp,
                      const bf16* __restrict__ Vt, const int* __restrict__ vlen,
                      bf16* __restrict__ O) {
  constexpr int SP = 1032;                // bf16 row stride (2064B, 16B-mult)
  extern __shared__ bf16 smem[];          // [32][SP]  (~66 KB)

  const int tid  = threadIdx.x;
  const int lane = tid & 31, wave = tid >> 5;
  const int ln = lane & 15, hi = lane >> 4;
  const int aoff = hi * 8, boff = hi * 8, coff = hi * 8;

  const int qt = blockIdx.x & 31;
  const int h  = (blockIdx.x >> 5) & 15;
  const int b  = blockIdx.x >> 9;
  const int q0 = qt * 32;
  const int vl = vlen[b];

  // ---- Phase 1: S = Q K^T / 8, mask, store bf16 to LDS -------------------
  {
    const int mt = wave & 1;
    const bf16* qrow = Qp + ((size_t)(b * Ss + q0 + mt * 16 + ln)) * Dd + h * DHh;
    Frag16 a0, a1;                        // d = 0..31 and 32..63
    a0.u[0] = *(const u32x4*)(qrow + aoff);
    a0.u[1] = *(const u32x4*)(qrow + 16 + aoff);
    a1.u[0] = *(const u32x4*)(qrow + 32 + aoff);
    a1.u[1] = *(const u32x4*)(qrow + 48 + aoff);

    for (int nt = (wave >> 1); nt < 64; nt += 4) {
      const bf16* krow = Kp + ((size_t)(b * Ss + nt * 16 + ln)) * Dd + h * DHh;
      Frag16 b0, b1;
      b0.u[0] = *(const u32x4*)(krow + boff);
      b0.u[1] = *(const u32x4*)(krow + boff + 16);
      b1.u[0] = *(const u32x4*)(krow + 32 + boff);
      b1.u[1] = *(const u32x4*)(krow + 32 + boff + 16);
      v8f c;
      #pragma unroll
      for (int r = 0; r < 8; ++r) c[r] = 0.0f;
      c = wmma_bf16(a0.v, b0.v, c);
      c = wmma_bf16(a1.v, b1.v, c);
      const int kv = nt * 16 + ln;
      #pragma unroll
      for (int r = 0; r < 8; ++r) {
        float val = (kv < vl) ? c[r] * 0.125f : -1.0e6f;  // scale 1/sqrt(64)
        smem[(mt * 16 + coff + r) * SP + kv] = (bf16)val;
      }
    }
  }
  __syncthreads();

  // ---- Phase 2: row softmax. Lane owns 4 chunks of 8 contiguous bf16 -----
  // (b128 LDS loads/stores; single read + single write per element)
  #pragma unroll
  for (int i = 0; i < 4; ++i) {
    const int row = wave * 4 + i;
    bf16* srow = smem + row * SP;
    Pack8 d[4];
    #pragma unroll
    for (int cc = 0; cc < 4; ++cc)
      d[cc].u = *(const u32x4*)(srow + cc * 256 + lane * 8);
    float vals[32];
    float mx = -3.0e38f;
    #pragma unroll
    for (int cc = 0; cc < 4; ++cc)
      #pragma unroll
      for (int j = 0; j < 8; ++j) {
        vals[cc * 8 + j] = (float)d[cc].h[j];
        mx = fmaxf(mx, vals[cc * 8 + j]);
      }
    mx = wave_max(mx);
    float sum = 0.0f;
    #pragma unroll
    for (int e = 0; e < 32; ++e) {
      vals[e] = __expf(vals[e] - mx);
      sum += vals[e];
    }
    sum = wave_sum(sum);
    const float inv = 1.0f / sum;
    #pragma unroll
    for (int cc = 0; cc < 4; ++cc) {
      #pragma unroll
      for (int j = 0; j < 8; ++j) d[cc].h[j] = (bf16)(vals[cc * 8 + j] * inv);
      *(u32x4*)(srow + cc * 256 + lane * 8) = d[cc].u;
    }
  }
  __syncthreads();

  // ---- Phase 3: O = P @ V  (A from LDS, B rows from transposed V) --------
  {
    const int mt = wave & 1;
    const int nt = wave >> 1;             // 0..3 -> 64 head dims
    v8f acc;
    #pragma unroll
    for (int r = 0; r < 8; ++r) acc[r] = 0.0f;
    const bf16* vrow = Vt + ((size_t)b * Dd + h * DHh + nt * 16 + ln) * (size_t)Ss;
    const bf16* prow = smem + (mt * 16 + ln) * SP;
    for (int kb = 0; kb < Ss; kb += 32) {
      Frag16 a, vv;
      a.u[0]  = *(const u32x4*)(prow + kb + aoff);
      a.u[1]  = *(const u32x4*)(prow + kb + 16 + aoff);
      vv.u[0] = *(const u32x4*)(vrow + kb + boff);
      vv.u[1] = *(const u32x4*)(vrow + kb + boff + 16);
      acc = wmma_bf16(a.v, vv.v, acc);
    }
    #pragma unroll
    for (int r = 0; r < 8; ++r)
      O[((size_t)(b * Ss + q0 + mt * 16 + coff + r)) * Dd +
        h * DHh + nt * 16 + ln] = (bf16)acc[r];
  }
}

// ---------------------------------------------------------------------------
extern "C" void kernel_launch(void* const* d_in, const int* in_sizes, int n_in,
                              void* d_out, int out_size, void* d_ws, size_t ws_size,
                              hipStream_t stream) {
  const float* q  = (const float*)d_in[0];
  const float* k  = (const float*)d_in[1];
  const float* v  = (const float*)d_in[2];
  const float* Wq = (const float*)d_in[3];
  const float* Wk = (const float*)d_in[4];
  const float* Wv = (const float*)d_in[5];
  const float* Wo = (const float*)d_in[6];
  const int* vl   = (const int*)d_in[7];

  char* ws = (char*)d_ws;
  size_t off = 0;
  const size_t wbytes = (size_t)Dd * Dd * sizeof(bf16);        // 2 MiB
  const size_t abytes = (size_t)Bb * Ss * Dd * sizeof(bf16);   // 16 MiB
  bf16* Wqt = (bf16*)(ws + off); off += wbytes;
  bf16* Wkt = (bf16*)(ws + off); off += wbytes;
  bf16* Wvt = (bf16*)(ws + off); off += wbytes;
  bf16* Wot = (bf16*)(ws + off); off += wbytes;
  bf16* Qp  = (bf16*)(ws + off); off += abytes;
  bf16* Kp  = (bf16*)(ws + off); off += abytes;
  bf16* Vtp = (bf16*)(ws + off); off += abytes;   // [B][1024][S] transposed
  bf16* Ob  = (bf16*)(ws + off); off += abytes;   // attention out, row-major

  const dim3 tb(256);
  const dim3 gT(Dd / 32, Dd / 32);
  transpose_to_bf16<<<gT, tb, 0, stream>>>(Wq, Wqt, Dd, Dd);
  transpose_to_bf16<<<gT, tb, 0, stream>>>(Wk, Wkt, Dd, Dd);
  transpose_to_bf16<<<gT, tb, 0, stream>>>(Wv, Wvt, Dd, Dd);
  transpose_to_bf16<<<gT, tb, 0, stream>>>(Wo, Wot, Dd, Dd);

  const int M = Bb * Ss;                   // 8192
  const dim3 gG(M / 128, Dd / 64);
  gemm_bf16_wmma<float, 0><<<gG, tb, 0, stream>>>(q, Wqt, Qp, M, Dd, Dd);
  gemm_bf16_wmma<float, 0><<<gG, tb, 0, stream>>>(k, Wkt, Kp, M, Dd, Dd);
  gemm_bf16_wmma<float, 1><<<gG, tb, 0, stream>>>(v, Wvt, Vtp, M, Dd, Dd);

  const size_t lds = (size_t)32 * 1032 * sizeof(bf16);  // 66048 B (<320 KB WGP)
  (void)hipFuncSetAttribute((const void*)attention_kernel,
                            hipFuncAttributeMaxDynamicSharedMemorySize, (int)lds);
  attention_kernel<<<dim3(Bb * Hh * (Ss / 32)), tb, lds, stream>>>(Qp, Kp, Vtp,
                                                                   vl, Ob);

  gemm_bf16_wmma<bf16, 2><<<gG, tb, 0, stream>>>(Ob, Wot, (float*)d_out,
                                                 M, Dd, Dd);
}
